// GCNDecoder_8306466750607
// MI455X (gfx1250) — compile-verified, matrix operand order checked by reading, not run on hardware
//
#include <hip/hip_runtime.h>
#include <hip/hip_bf16.h>

// ---------------- problem constants (B=8, N=2048, D=H=256) ----------------
#define PB 8
#define PN 2048
#define PD 256

typedef __attribute__((ext_vector_type(16))) _Float16 v16h;
typedef __attribute__((ext_vector_type(8)))  _Float16 v8h;
typedef __attribute__((ext_vector_type(8)))  float    v8f;

// ---------------- WMMA helpers ----------------
__device__ __forceinline__ v8f wmma32(v16h a, v16h b, v8f c) {
    // D = A(16x32 f16) * B(32x16 f16) + C(16x16 f32)
    return __builtin_amdgcn_wmma_f32_16x16x32_f16(
        /*neg_a=*/false, a, /*neg_b=*/false, b,
        /*c_mod=*/(short)0, c, /*reuse_a=*/false, /*reuse_b=*/false);
}

// A-frag (16xK slice, row-major source: lane -> row, halves -> K) or
// B-frag from a row-major matrix M when B = M^T (lane -> column of B == row of M).
// Half h of lane L maps to K = (L>=16 ? 8:0) + (h<8 ? h : h+8)  [ISA 7.12.2].
// Two contiguous 8-half (16B) chunks per lane -> pairs of b128 loads.
__device__ __forceinline__ v16h frag_ld(const _Float16* p, int row0, int ld,
                                        int k0, int lane) {
    const int hi = (lane >> 4) & 1;
    const _Float16* base = p + (size_t)(row0 + (lane & 15)) * ld + k0 + hi * 8;
    v16h f;
#pragma unroll
    for (int i = 0; i < 8; ++i) { f[i] = base[i]; f[i + 8] = base[i + 16]; }
    return f;
}

__device__ __forceinline__ float sigf(float x) {
    // one v_exp + one v_rcp (no IEEE-divide Newton chain)
    return __builtin_amdgcn_rcpf(1.0f + __expf(-x));
}

// ---- 1) normalize rows, cast to f16 (row-major Xn, transposed Xt), zero deg
__global__ __launch_bounds__(256) void k_prep(const float* __restrict__ X,
                                              _Float16* __restrict__ Xn16,
                                              _Float16* __restrict__ Xt16,
                                              float* __restrict__ deg) {
    const int row  = blockIdx.x * 8 + (threadIdx.x >> 5);   // over B*N
    const int lane = threadIdx.x & 31;
    const float* xr = X + (size_t)row * PD + lane * 8;
    float v[8];
    float ss = 0.0f;
#pragma unroll
    for (int i = 0; i < 8; ++i) { v[i] = xr[i]; ss += v[i] * v[i]; }
#pragma unroll
    for (int d = 1; d < 32; d <<= 1) ss += __shfl_xor(ss, d, 32);
    const float inv = 1.0f / fmaxf(sqrtf(ss), 1e-12f);
    _Float16* pn = Xn16 + (size_t)row * PD + lane * 8;
#pragma unroll
    for (int i = 0; i < 8; ++i) pn[i] = (_Float16)(v[i] * inv);
    // transposed copy: Xt[b][d][n] = X[b][n][d]
    const int b = row >> 11;
    const int n = row & (PN - 1);
    _Float16* pt = Xt16 + ((size_t)b * PD + lane * 8) * PN + n;
#pragma unroll
    for (int i = 0; i < 8; ++i) pt[(size_t)i * PN] = (_Float16)v[i];
    if (lane == 0) deg[row] = 0.0f;
}

// ---- 2) W1/W2 -> transposed f16 (Wt[n][k] = W[k][n]) ----
__global__ __launch_bounds__(256) void k_convw(const float* __restrict__ W1,
                                               const float* __restrict__ W2,
                                               _Float16* __restrict__ W1t,
                                               _Float16* __restrict__ W2t) {
    const int i = blockIdx.x * 256 + threadIdx.x;           // 65536 each
    const int k = i >> 8, n = i & 255;
    W1t[n * 256 + k] = (_Float16)W1[i];
    W2t[n * 256 + k] = (_Float16)W2[i];
}

// ---- 3) degree pass; optionally cache S_sig = sig*mi*mj(+diag) to L2 scratch
__global__ __launch_bounds__(256) void k_deg(const _Float16* __restrict__ Xn16,
                                             const int* __restrict__ mask,
                                             float* __restrict__ deg,
                                             _Float16* __restrict__ Sws) {
    const int bid   = blockIdx.x;           // 8 jblk * 128 itile * 8 b
    const int jblk  = bid & 7;
    const int itile = (bid >> 3) & 127;
    const int b     = bid >> 10;
    const int wave  = threadIdx.x >> 5;
    const int lane  = threadIdx.x & 31;
    const int hi    = (lane >> 4) & 1;
    const int i0    = itile * 16;
    const size_t bN = (size_t)b * PN;
    const _Float16* Xn = Xn16 + bN * PD;

    float mi[8];
#pragma unroll
    for (int r = 0; r < 8; ++r) mi[r] = (float)mask[bN + i0 + r + 8 * hi];

    v16h afr[8];                             // preload i-tile A-frags once
#pragma unroll
    for (int kk = 0; kk < 8; ++kk) afr[kk] = frag_ld(Xn, i0, PD, kk * 32, lane);

#pragma unroll
    for (int t = 0; t < 2; ++t) {
        const int j0 = jblk * 256 + (wave * 2 + t) * 16;
        const int gj = j0 + (lane & 15);
        const float mj = (float)mask[bN + gj];
        v8f c = {};
#pragma unroll
        for (int kk = 0; kk < 8; ++kk)
            c = wmma32(afr[kk], frag_ld(Xn, j0, PD, kk * 32, lane), c);
#pragma unroll
        for (int r = 0; r < 8; ++r) {
            const float sg = sigf(c[r]);
            if (Sws) {                      // masked + self-loop, pre-D-norm
                float st = sg * mi[r] * mj;
                if (i0 + r + 8 * hi == gj) st += mi[r];
                Sws[(bN + i0 + r + 8 * hi) * PN + gj] = (_Float16)st;
            }
            float s = sg * mj;              // row-sum term (m_i folded in k_fin)
#pragma unroll
            for (int d = 1; d < 16; d <<= 1) s += __shfl_xor(s, d, 32);
            if ((lane & 15) == 0)
                atomicAdd(&deg[bN + i0 + r + 8 * hi], s);
        }
    }
}

// ---- 4) d_i = rsqrt(max(m_i*(deg_i+1), 1e-6)) ----
__global__ __launch_bounds__(256) void k_fin(float* __restrict__ deg,
                                             const int* __restrict__ mask) {
    const int i = blockIdx.x * 256 + threadIdx.x;           // B*N = 16384
    const float m = (float)mask[i];
    deg[i] = rsqrtf(fmaxf(m * (deg[i] + 1.0f), 1e-6f));
}

// ---- 5) agg = D^-1/2 (A+I_m) D^-1/2 @ X, fused via LDS ----
// Sws != nullptr: phase 1 streams cached S_sig from L2 and applies d_i*d_j.
// Sws == nullptr: phase 1 recomputes S via WMMA (fallback if ws is small).
__global__ __launch_bounds__(256) void k_agg(const _Float16* __restrict__ Xn16,
                                             const _Float16* __restrict__ Xt16,
                                             const int* __restrict__ mask,
                                             const float* __restrict__ dinv,
                                             const _Float16* __restrict__ Sws,
                                             _Float16* __restrict__ agg16) {
    __shared__ _Float16 sS[16 * PN];        // 64 KB: one 16-row A_norm slice
    const int b     = blockIdx.x >> 7;
    const int itile = blockIdx.x & 127;
    const int i0    = itile * 16;
    const int wave  = threadIdx.x >> 5;
    const int lane  = threadIdx.x & 31;
    const int hi    = (lane >> 4) & 1;
    const size_t bN = (size_t)b * PN;
    const _Float16* Xn = Xn16 + bN * PD;
    const _Float16* Xt = Xt16 + bN * PD;    // [PD][PN] transposed slice

    if (Sws) {
        // cached phase 1: vectorized scale-copy, L2 -> LDS
#pragma unroll
        for (int it = 0; it < 16; ++it) {
            const int e0 = ((int)threadIdx.x + it * 256) * 8;  // 16*2048 elems
            const int m = e0 >> 11, j = e0 & (PN - 1);
            const float dim = dinv[bN + i0 + m];
            const v8h sv = *(const v8h*)(Sws + (bN + i0 + m) * PN + j);
            v8h ov;
#pragma unroll
            for (int q = 0; q < 8; ++q)
                ov[q] = (_Float16)((float)sv[q] * dim * dinv[bN + j + q]);
            *(v8h*)(sS + m * PN + j) = ov;
        }
    } else {
        float mi[8], di[8];
#pragma unroll
        for (int r = 0; r < 8; ++r) {
            const int gi = i0 + r + 8 * hi;
            mi[r] = (float)mask[bN + gi];
            di[r] = dinv[bN + gi];
        }
        v16h afr[8];                         // i-tile Xn A-frags, reused 16x
#pragma unroll
        for (int kk = 0; kk < 8; ++kk)
            afr[kk] = frag_ld(Xn, i0, PD, kk * 32, lane);
        for (int jt = wave; jt < PN / 16; jt += 8) {
            const int j0 = jt * 16;
            const int gj = j0 + (lane & 15);
            const float mj = (float)mask[bN + gj];
            const float dj = dinv[bN + gj];
            v8f c = {};
#pragma unroll
            for (int kk = 0; kk < 8; ++kk)
                c = wmma32(afr[kk], frag_ld(Xn, j0, PD, kk * 32, lane), c);
#pragma unroll
            for (int r = 0; r < 8; ++r) {
                const int m = r + 8 * hi;
                float s = sigf(c[r]) * mi[r] * mj;
                if (i0 + m == gj) s += mi[r];
                s *= di[r] * dj;
                sS[m * PN + gj] = (_Float16)s;
            }
        }
    }
    __syncthreads();

    // phase 2: agg tile = sS(16x2048) @ X(2048x256); B-frags from Xt (b128)
#pragma unroll
    for (int t = 0; t < 2; ++t) {
        const int n0 = (wave * 2 + t) * 16;
        v8f c = {};
        for (int k0 = 0; k0 < PN; k0 += 32) {
            v16h a  = frag_ld(sS, 0, PN, k0, lane);        // ds_load
            v16h bf = frag_ld(Xt, n0, PN, k0, lane);       // global b128
            c = wmma32(a, bf, c);
        }
#pragma unroll
        for (int r = 0; r < 8; ++r) {
            const int m = r + 8 * hi;
            agg16[(bN + i0 + m) * PD + n0 + (lane & 15)] = (_Float16)c[r];
        }
    }
}

// ---- 6) P = relu(agg@W1+b1)@W2+b2, Hf staged in LDS ----
__global__ __launch_bounds__(256) void k_mlp(const _Float16* __restrict__ agg16,
                                             const _Float16* __restrict__ W1t,
                                             const float* __restrict__ b1,
                                             const _Float16* __restrict__ W2t,
                                             const float* __restrict__ b2,
                                             _Float16* __restrict__ P16) {
    __shared__ _Float16 sH[16 * PD];        // 8 KB
    const int b     = blockIdx.x >> 7;
    const int itile = blockIdx.x & 127;
    const int i0    = itile * 16;
    const int wave  = threadIdx.x >> 5;
    const int lane  = threadIdx.x & 31;
    const int hi    = (lane >> 4) & 1;
    const size_t bN = (size_t)b * PN;
    const _Float16* Arow = agg16 + (bN + i0) * PD;

    v16h afr[8];
#pragma unroll
    for (int kk = 0; kk < 8; ++kk) afr[kk] = frag_ld(Arow, 0, PD, kk * 32, lane);

#pragma unroll
    for (int t = 0; t < 2; ++t) {
        const int n0 = (wave * 2 + t) * 16;
        const float bias = b1[n0 + (lane & 15)];
        v8f c = {};
#pragma unroll
        for (int kk = 0; kk < 8; ++kk)
            c = wmma32(afr[kk], frag_ld(W1t, n0, PD, kk * 32, lane), c);
#pragma unroll
        for (int r = 0; r < 8; ++r)
            sH[(r + 8 * hi) * PD + n0 + (lane & 15)] =
                (_Float16)fmaxf(c[r] + bias, 0.0f);
    }
    __syncthreads();

    v16h hfr[8];                             // Hf A-frags from LDS, reused
#pragma unroll
    for (int kk = 0; kk < 8; ++kk) hfr[kk] = frag_ld(sH, 0, PD, kk * 32, lane);

#pragma unroll
    for (int t = 0; t < 2; ++t) {
        const int n0 = (wave * 2 + t) * 16;
        const float bias = b2[n0 + (lane & 15)];
        v8f c = {};
#pragma unroll
        for (int kk = 0; kk < 8; ++kk)
            c = wmma32(hfr[kk], frag_ld(W2t, n0, PD, kk * 32, lane), c);
#pragma unroll
        for (int r = 0; r < 8; ++r)
            P16[(bN + i0 + r + 8 * hi) * PD + n0 + (lane & 15)] =
                (_Float16)(c[r] + bias);
    }
}

// ---- 7) out = pair_mask(sigmoid(P @ P^T)) ----
__global__ __launch_bounds__(256) void k_dec(const _Float16* __restrict__ P16,
                                             const int* __restrict__ mask,
                                             float* __restrict__ out) {
    const int bid   = blockIdx.x;           // 8 jblk * 128 itile * 8 b
    const int jblk  = bid & 7;
    const int itile = (bid >> 3) & 127;
    const int b     = bid >> 10;
    const int wave  = threadIdx.x >> 5;
    const int lane  = threadIdx.x & 31;
    const int hi    = (lane >> 4) & 1;
    const int i0    = itile * 16;
    const size_t bN = (size_t)b * PN;
    const _Float16* Pb = P16 + bN * PD;

    float mi[8];
#pragma unroll
    for (int r = 0; r < 8; ++r) mi[r] = (float)mask[bN + i0 + r + 8 * hi];

    v16h afr[8];
#pragma unroll
    for (int kk = 0; kk < 8; ++kk) afr[kk] = frag_ld(Pb, i0, PD, kk * 32, lane);

#pragma unroll
    for (int t = 0; t < 2; ++t) {
        const int j0 = jblk * 256 + (wave * 2 + t) * 16;
        const float mj = (float)mask[bN + j0 + (lane & 15)];
        v8f c = {};
#pragma unroll
        for (int kk = 0; kk < 8; ++kk)
            c = wmma32(afr[kk], frag_ld(Pb, j0, PD, kk * 32, lane), c);
#pragma unroll
        for (int r = 0; r < 8; ++r) {
            const int m = r + 8 * hi;
            out[((size_t)b * PN + i0 + m) * PN + j0 + (lane & 15)] =
                sigf(c[r]) * mi[r] * mj;
        }
    }
}

// ---------------- launcher ----------------
extern "C" void kernel_launch(void* const* d_in, const int* in_sizes, int n_in,
                              void* d_out, int out_size, void* d_ws, size_t ws_size,
                              hipStream_t stream) {
    const float* X    = (const float*)d_in[0];
    const int*   mask = (const int*)d_in[1];
    const float* W1   = (const float*)d_in[2];
    const float* b1   = (const float*)d_in[3];
    const float* W2   = (const float*)d_in[4];
    const float* b2   = (const float*)d_in[5];
    float* out = (float*)d_out;

    char* ws = (char*)d_ws;
    const size_t feat16 = (size_t)PB * PN * PD * sizeof(_Float16); // 8 MB
    const size_t wmat   = 256 * 256 * sizeof(_Float16);
    const size_t degsz  = (size_t)PB * PN * sizeof(float);
    _Float16* Xn16  = (_Float16*)(ws);
    _Float16* Xt16  = (_Float16*)(ws + feat16);
    _Float16* agg16 = (_Float16*)(ws + 2 * feat16);
    _Float16* P16   = (_Float16*)(ws + 3 * feat16);
    _Float16* W1t   = (_Float16*)(ws + 4 * feat16);
    _Float16* W2t   = (_Float16*)(ws + 4 * feat16 + wmat);
    float*    deg   = (float*)   (ws + 4 * feat16 + 2 * wmat);
    const size_t base  = 4 * feat16 + 2 * wmat + degsz;
    const size_t s_bytes = (size_t)PB * PN * PN * sizeof(_Float16); // 64 MB
    // cache S_sig in scratch (L2-resident) when workspace permits
    _Float16* Sws = (ws_size >= base + s_bytes) ? (_Float16*)(ws + base) : nullptr;

    k_prep <<<PB * PN / 8, 256, 0, stream>>>(X, Xn16, Xt16, deg);
    k_convw<<<256, 256, 0, stream>>>(W1, W2, W1t, W2t);
    k_deg  <<<PB * (PN / 16) * (PN / 256), 256, 0, stream>>>(Xn16, mask, deg, Sws);
    k_fin  <<<PB * PN / 256, 256, 0, stream>>>(deg, mask);
    k_agg  <<<PB * (PN / 16), 256, 0, stream>>>(Xn16, Xt16, mask, deg, Sws, agg16);
    k_mlp  <<<PB * (PN / 16), 256, 0, stream>>>(agg16, W1t, b1, W2t, b2, P16);
    k_dec  <<<PB * (PN / 16) * (PN / 256), 256, 0, stream>>>(P16, mask, out);
    (void)in_sizes; (void)n_in; (void)out_size; (void)ws_size;
}
